// TransformerMIDILanguageModel_4028679324191
// MI455X (gfx1250) — matrix operand order repaired and checked by reading, not compiled, not run on hardware
//
#include <hip/hip_runtime.h>
#include <hip/hip_bf16.h>

#define L_  6
#define H_  8
#define C_  512
#define HD_ 64
#define T_  1024
#define V_  512
#define B_  4
#define FF_ (4 * C_)

typedef __attribute__((ext_vector_type(16))) __bf16 v16bf;
typedef __attribute__((ext_vector_type(8)))  __bf16 bf16x8;
typedef __attribute__((ext_vector_type(4)))  __bf16 bf16x4;
typedef __attribute__((ext_vector_type(8)))  float  v8f;

// ---------------------------------------------------------------------------
// Embedding gather: x[b,t,:] = emb[idx[b,t],:]
// ---------------------------------------------------------------------------
__global__ void embed_kernel(const int* __restrict__ idx,
                             const float* __restrict__ emb,
                             float* __restrict__ x) {
  int row = blockIdx.x;                 // 0 .. B*T-1
  int id  = idx[row];
  const float* src = emb + (size_t)id * C_;
  float* dst = x + (size_t)row * C_;
  for (int c = threadIdx.x; c < C_; c += blockDim.x) dst[c] = src[c];
}

// ---------------------------------------------------------------------------
// Row LayerNorm over C columns, one block per row
// ---------------------------------------------------------------------------
__global__ void layernorm_kernel(const float* __restrict__ in,
                                 float* __restrict__ out,
                                 const float* __restrict__ g,
                                 const float* __restrict__ b,
                                 int C) {
  __shared__ float s1[256];
  __shared__ float s2[256];
  int row = blockIdx.x;
  int tid = threadIdx.x;
  const float* p = in + (size_t)row * C;
  float sum = 0.f, sq = 0.f;
  for (int c = tid; c < C; c += 256) { float v = p[c]; sum += v; sq += v * v; }
  s1[tid] = sum; s2[tid] = sq; __syncthreads();
  for (int s = 128; s > 0; s >>= 1) {
    if (tid < s) { s1[tid] += s1[tid + s]; s2[tid] += s2[tid + s]; }
    __syncthreads();
  }
  float mean = s1[0] / (float)C;
  float var  = s2[0] / (float)C - mean * mean;
  float r = rsqrtf(var + 1e-5f);
  float* o = out + (size_t)row * C;
  for (int c = tid; c < C; c += 256) o[c] = (p[c] - mean) * r * g[c] + b[c];
}

// ---------------------------------------------------------------------------
// Repack Wq/Wk/Wv layer l: (H,C,HD) -> (C, H*HD) so QKV is one GEMM each
// ---------------------------------------------------------------------------
__global__ void pack_qkv_kernel(const float* __restrict__ Wq,
                                const float* __restrict__ Wk,
                                const float* __restrict__ Wv,
                                float* __restrict__ pq,
                                float* __restrict__ pk,
                                float* __restrict__ pv, int l) {
  int i = blockIdx.x * blockDim.x + threadIdx.x;   // over C*H*HD
  int d = i % HD_;
  int c = (i / HD_) % C_;
  int h = i / (HD_ * C_);
  int dst = c * (H_ * HD_) + h * HD_ + d;
  size_t src = (((size_t)l * H_ + h) * C_ + c) * HD_ + d;
  pq[dst] = Wq[src]; pk[dst] = Wk[src]; pv[dst] = Wv[src];
}

// ---------------------------------------------------------------------------
// Row softmax (length T), one block per (b,h,t) row, in place.
// ---------------------------------------------------------------------------
__global__ void softmax_kernel(float* __restrict__ att) {
  __shared__ float sm[256];
  size_t base = (size_t)blockIdx.x * T_;
  int tid = threadIdx.x;
  float mx = -__builtin_inff();
  for (int j = tid; j < T_; j += 256) mx = fmaxf(mx, att[base + j]);
  sm[tid] = mx; __syncthreads();
  for (int s = 128; s > 0; s >>= 1) {
    if (tid < s) sm[tid] = fmaxf(sm[tid], sm[tid + s]);
    __syncthreads();
  }
  mx = sm[0]; __syncthreads();
  float sum = 0.f;
  for (int j = tid; j < T_; j += 256) {
    float e = expf(att[base + j] - mx);
    att[base + j] = e; sum += e;
  }
  sm[tid] = sum; __syncthreads();
  for (int s = 128; s > 0; s >>= 1) {
    if (tid < s) sm[tid] += sm[tid + s];
    __syncthreads();
  }
  float inv = 1.0f / sm[0];
  for (int j = tid; j < T_; j += 256) att[base + j] *= inv;
}

// ---------------------------------------------------------------------------
// Batched WMMA GEMM.  D(MxN) = epilogue(A(MxK) @ B(KxN or NxK^T))
// Block tile 64x64, 256 threads = 8 waves; each wave owns a 32x16 slab
// (two 16x16 f32 accumulators sharing one B fragment).
// fp32 operands are staged to LDS as bf16 tiles with layouts chosen so every
// WMMA fragment is contiguous 16B runs -> ds_load_b128 (ISA 7.12.2):
//   A row-major  lsA[m][k]  (stride 40): lane run k = half*8+{0..7}, 16+half*8+{0..7}
//   B transposed lsBt[n][k] (stride 40): lane run k = half*8+{0..7}, 16+half*8+{0..7}
// Core op: v_wmma_f32_16x16x32_bf16.
// Batch (grid.z): z -> b=z/H_, h=z%H_; per-matrix base offsets b*s?b + h*s?h.
// EPI: 0 none | 1 +bias | 2 res + acc + bias | 3 gelu(acc+bias)
//      4 attention logits: acc*scale + skew(aux), causal mask (-inf)
// ---------------------------------------------------------------------------
#define MT 64
#define NT 64
#define KT 32
#define LSTR 40   // padded LDS K-stride (elements): rows 16B aligned, banks spread

template <int EPI, bool BTR>
__global__ void __launch_bounds__(256)
gemm_wmma(const float* __restrict__ A, const float* __restrict__ Bm,
          float* __restrict__ D,
          const float* __restrict__ bias, const float* __restrict__ res,
          const float* __restrict__ aux,
          int M, int N, int K, int lda, int ldb, int ldc,
          long sAb, long sAh, long sBb, long sBh,
          long sDb, long sDh, long sXb, long sXh,
          float scale) {
  __shared__ __bf16 lsA[MT * LSTR];   // [m][k]
  __shared__ __bf16 lsB[NT * LSTR];   // [n][k]  (transposed tile)

  int z  = blockIdx.z;
  int zb = z / H_, zh = z % H_;
  A  += zb * sAb + zh * sAh;
  Bm += zb * sBb + zh * sBh;
  D  += zb * sDb + zh * sDh;
  const float* auxp = aux + zb * sXb + zh * sXh;

  int m0 = blockIdx.y * MT;
  int n0 = blockIdx.x * NT;
  int tid  = threadIdx.x;
  int lane = tid & 31;
  int wave = tid >> 5;
  int wmg = (wave & 1) * 32;  // 32-row slab base
  int wn  = (wave >> 1) * 16; // 16-col subtile base
  int half = lane >> 4;
  int l16  = lane & 15;

  v8f acc0 = {};
  v8f acc1 = {};

  for (int k0 = 0; k0 < K; k0 += KT) {
    // ---- stage A tile (MT x KT): float4 global loads -> bf16x4 LDS stores
#pragma unroll
    for (int it = 0; it < (MT * KT / 4) / 256; ++it) {
      int ci = tid + it * 256;            // chunk of 4 along K
      int r  = ci >> 3;
      int c4 = (ci & 7) << 2;
      float4 f = *(const float4*)&A[(size_t)(m0 + r) * lda + k0 + c4];
      bf16x4 bv = {(__bf16)f.x, (__bf16)f.y, (__bf16)f.z, (__bf16)f.w};
      *(bf16x4*)&lsA[r * LSTR + c4] = bv;
    }
    // ---- stage B tile transposed to lsB[n][k]
    if constexpr (BTR) {
      // source is (N x K) row-major: contiguous along k
#pragma unroll
      for (int it = 0; it < (NT * KT / 4) / 256; ++it) {
        int ci = tid + it * 256;
        int n  = ci >> 3;
        int c4 = (ci & 7) << 2;
        float4 f = *(const float4*)&Bm[(size_t)(n0 + n) * ldb + k0 + c4];
        bf16x4 bv = {(__bf16)f.x, (__bf16)f.y, (__bf16)f.z, (__bf16)f.w};
        *(bf16x4*)&lsB[n * LSTR + c4] = bv;
      }
    } else {
      // source is (K x N) row-major: contiguous along n, scatter into lsB
#pragma unroll
      for (int it = 0; it < (KT * NT / 4) / 256; ++it) {
        int ci = tid + it * 256;
        int k  = ci >> 4;
        int n4 = (ci & 15) << 2;
        float4 f = *(const float4*)&Bm[(size_t)(k0 + k) * ldb + n0 + n4];
        lsB[(n4 + 0) * LSTR + k] = (__bf16)f.x;
        lsB[(n4 + 1) * LSTR + k] = (__bf16)f.y;
        lsB[(n4 + 2) * LSTR + k] = (__bf16)f.z;
        lsB[(n4 + 3) * LSTR + k] = (__bf16)f.w;
      }
    }
    // prefetch next K tile into cache while we compute
    if (k0 + KT < K) {
      __builtin_prefetch(&A[(size_t)(m0 + (tid & 63)) * lda + k0 + KT], 0, 1);
      if constexpr (BTR)
        __builtin_prefetch(&Bm[(size_t)(n0 + (tid & 63)) * ldb + k0 + KT], 0, 1);
      else
        __builtin_prefetch(&Bm[(size_t)(k0 + KT + (tid & 31)) * ldb + n0], 0, 1);
    }
    __syncthreads();

    // ---- fragments: contiguous ds_load_b128 pairs
    int nn = wn + l16;
    bf16x8 b0 = *(const bf16x8*)&lsB[nn * LSTR + half * 8];
    bf16x8 b1 = *(const bf16x8*)&lsB[nn * LSTR + 16 + half * 8];
    v16bf bfr = __builtin_shufflevector(b0, b1, 0, 1, 2, 3, 4, 5, 6, 7, 8, 9,
                                        10, 11, 12, 13, 14, 15);
    int mm0 = wmg + l16;
    bf16x8 a0 = *(const bf16x8*)&lsA[mm0 * LSTR + half * 8];
    bf16x8 a1 = *(const bf16x8*)&lsA[mm0 * LSTR + 16 + half * 8];
    v16bf af0 = __builtin_shufflevector(a0, a1, 0, 1, 2, 3, 4, 5, 6, 7, 8, 9,
                                        10, 11, 12, 13, 14, 15);
    int mm1 = mm0 + 16;
    bf16x8 a2 = *(const bf16x8*)&lsA[mm1 * LSTR + half * 8];
    bf16x8 a3 = *(const bf16x8*)&lsA[mm1 * LSTR + 16 + half * 8];
    v16bf af1 = __builtin_shufflevector(a2, a3, 0, 1, 2, 3, 4, 5, 6, 7, 8, 9,
                                        10, 11, 12, 13, 14, 15);

    acc0 = __builtin_amdgcn_wmma_f32_16x16x32_bf16(false, af0, false, bfr,
                                                   (short)0, acc0, false, false);
    acc1 = __builtin_amdgcn_wmma_f32_16x16x32_bf16(false, af1, false, bfr,
                                                   (short)0, acc1, false, false);
    __syncthreads();
  }

  // ---- epilogue: C/D layout -> lane l16 = N, VGPR r = M (+8 upper lane half)
  int gn = n0 + wn + l16;
#pragma unroll
  for (int sub = 0; sub < 2; ++sub) {
    v8f acc = sub ? acc1 : acc0;
#pragma unroll
    for (int r = 0; r < 8; ++r) {
      int gm = m0 + wmg + sub * 16 + half * 8 + r;
      size_t o = (size_t)gm * ldc + gn;
      float v = acc[r];
      if constexpr (EPI == 1) {
        v += bias[gn];
      } else if constexpr (EPI == 2) {
        v = res[o] + v + bias[gn];
      } else if constexpr (EPI == 3) {
        v += bias[gn];
        v = 0.5f * v * (1.0f + erff(v * 0.70710678118654752f));  // exact GELU
      } else if constexpr (EPI == 4) {
        // reference _skew: out[i,j] <- padded flat (j+1)*T+i, rel row len T+1
        int i = gm, j = gn;
        int flat = (j + 1) * T_ + i;
        int tt = flat / (T_ + 1);
        int ss = flat % (T_ + 1);
        float sk = (ss == 0) ? 0.0f : auxp[(size_t)tt * T_ + (ss - 1)];
        v = v * scale + sk;
        if (j > i) v = -__builtin_inff();                        // causal mask
      }
      D[o] = v;
    }
  }
}

// ---------------------------------------------------------------------------
extern "C" void kernel_launch(void* const* d_in, const int* in_sizes, int n_in,
                              void* d_out, int out_size, void* d_ws,
                              size_t ws_size, hipStream_t stream) {
  (void)in_sizes; (void)n_in; (void)out_size; (void)ws_size;

  const int*   idx  = (const int*)  d_in[0];
  const float* emb  = (const float*)d_in[1];
  const float* Wq   = (const float*)d_in[2];
  const float* Wk   = (const float*)d_in[3];
  const float* Wv   = (const float*)d_in[4];
  const float* Er   = (const float*)d_in[5];
  const float* Wo   = (const float*)d_in[6];
  const float* bo   = (const float*)d_in[7];
  const float* ln1g = (const float*)d_in[8];
  const float* ln1b = (const float*)d_in[9];
  const float* ln2g = (const float*)d_in[10];
  const float* ln2b = (const float*)d_in[11];
  const float* W1   = (const float*)d_in[12];
  const float* b1   = (const float*)d_in[13];
  const float* W2   = (const float*)d_in[14];
  const float* b2   = (const float*)d_in[15];
  const float* lnfg = (const float*)d_in[16];
  const float* lnfb = (const float*)d_in[17];
  const float* Wlm  = (const float*)d_in[18];
  const float* blm  = (const float*)d_in[19];

  const size_t NR = (size_t)B_ * T_;   // 4096 rows
  float* w = (float*)d_ws;
  size_t off = 0;
  auto alloc = [&](size_t n) { float* p = w + off; off += n; return p; };
  float* xbuf   = alloc(NR * C_);
  float* hbuf   = alloc(NR * C_);
  float* qbuf   = alloc(NR * C_);       // (B,T,H*HD) == (B,H,T,HD) via strides
  float* kbuf   = alloc(NR * C_);
  float* vbuf   = alloc(NR * C_);
  float* obuf   = alloc(NR * C_);
  float* ffbuf  = alloc(NR * FF_);
  float* pq     = alloc((size_t)C_ * C_);
  float* pk     = alloc((size_t)C_ * C_);
  float* pv     = alloc((size_t)C_ * C_);
  float* relbuf = alloc((size_t)B_ * H_ * T_ * T_);
  float* attbuf = alloc((size_t)B_ * H_ * T_ * T_);

  embed_kernel<<<(int)NR, 256, 0, stream>>>(idx, emb, xbuf);

  const long sQb = (long)T_ * C_, sQh = HD_;                 // q/k/v/o per-(b,h)
  const long sTb = (long)H_ * T_ * T_, sTh = (long)T_ * T_;  // att/rel per-(b,h)
  const dim3 blk(256);
  const dim3 gCC(C_ / NT, (unsigned)(NR / MT), 1);       // (M=4096, N=512)
  const dim3 gAtt(T_ / NT, T_ / MT, B_ * H_);            // (M=1024, N=1024)
  const dim3 gAV(HD_ / NT, T_ / MT, B_ * H_);            // (M=1024, N=64)
  const dim3 gFF(FF_ / NT, (unsigned)(NR / MT), 1);      // (M=4096, N=2048)

  for (int l = 0; l < L_; ++l) {
    layernorm_kernel<<<(int)NR, blk, 0, stream>>>(xbuf, hbuf, ln1g + l * C_,
                                                  ln1b + l * C_, C_);
    pack_qkv_kernel<<<(C_ * H_ * HD_) / 256, blk, 0, stream>>>(Wq, Wk, Wv, pq,
                                                               pk, pv, l);
    // q,k,v = h @ Wpack   (stored as (B,T,H*HD))
    gemm_wmma<0, false><<<gCC, blk, 0, stream>>>(
        hbuf, pq, qbuf, nullptr, nullptr, nullptr, (int)NR, C_, C_, C_, C_, C_,
        0, 0, 0, 0, 0, 0, 0, 0, 1.0f);
    gemm_wmma<0, false><<<gCC, blk, 0, stream>>>(
        hbuf, pk, kbuf, nullptr, nullptr, nullptr, (int)NR, C_, C_, C_, C_, C_,
        0, 0, 0, 0, 0, 0, 0, 0, 1.0f);
    gemm_wmma<0, false><<<gCC, blk, 0, stream>>>(
        hbuf, pv, vbuf, nullptr, nullptr, nullptr, (int)NR, C_, C_, C_, C_, C_,
        0, 0, 0, 0, 0, 0, 0, 0, 1.0f);
    // rel = q @ Er[l,h]^T     (batched over b,h; Er independent of b)
    gemm_wmma<0, true><<<gAtt, blk, 0, stream>>>(
        qbuf, Er + (size_t)l * H_ * T_ * HD_, relbuf, nullptr, nullptr, nullptr,
        T_, T_, HD_, C_, HD_, T_,
        sQb, sQh, 0, (long)T_ * HD_, sTb, sTh, 0, 0, 1.0f);
    // att = mask(q @ k^T * 1/sqrt(HD) + skew(rel))
    gemm_wmma<4, true><<<gAtt, blk, 0, stream>>>(
        qbuf, kbuf, attbuf, nullptr, nullptr, relbuf,
        T_, T_, HD_, C_, C_, T_,
        sQb, sQh, sQb, sQh, sTb, sTh, sTb, sTh, 0.125f);
    softmax_kernel<<<B_ * H_ * T_, blk, 0, stream>>>(attbuf);
    // o = att @ v  (written straight into (B,T,H*HD) => heads concatenated)
    gemm_wmma<0, false><<<gAV, blk, 0, stream>>>(
        attbuf, vbuf, obuf, nullptr, nullptr, nullptr, T_, HD_, T_, T_, C_, C_,
        sTb, sTh, sQb, sQh, sQb, sQh, 0, 0, 1.0f);
    // x = x + o @ Wo + bo
    gemm_wmma<2, false><<<gCC, blk, 0, stream>>>(
        obuf, Wo + (size_t)l * C_ * C_, xbuf, bo + l * C_, xbuf, nullptr,
        (int)NR, C_, C_, C_, C_, C_, 0, 0, 0, 0, 0, 0, 0, 0, 1.0f);
    // FFN
    layernorm_kernel<<<(int)NR, blk, 0, stream>>>(xbuf, hbuf, ln2g + l * C_,
                                                  ln2b + l * C_, C_);
    gemm_wmma<3, false><<<gFF, blk, 0, stream>>>(
        hbuf, W1 + (size_t)l * C_ * FF_, ffbuf, b1 + l * FF_, nullptr, nullptr,
        (int)NR, FF_, C_, C_, FF_, FF_, 0, 0, 0, 0, 0, 0, 0, 0, 1.0f);
    gemm_wmma<2, false><<<gCC, blk, 0, stream>>>(
        ffbuf, W2 + (size_t)l * FF_ * C_, xbuf, b2 + l * C_, xbuf, nullptr,
        (int)NR, C_, FF_, FF_, C_, C_, 0, 0, 0, 0, 0, 0, 0, 0, 1.0f);
  }

  layernorm_kernel<<<(int)NR, blk, 0, stream>>>(xbuf, hbuf, lnfg, lnfb, C_);
  gemm_wmma<1, false><<<dim3(V_ / NT, (unsigned)(NR / MT), 1), blk, 0, stream>>>(
      hbuf, Wlm, (float*)d_out, blm, nullptr, nullptr, (int)NR, V_, C_, C_, V_,
      V_, 0, 0, 0, 0, 0, 0, 0, 0, 1.0f);
}
